// WindowAttention_30116310680108
// MI455X (gfx1250) — matrix operand order repaired and checked by reading, not compiled, not run on hardware
//
#include <hip/hip_runtime.h>

typedef _Float16 h16;
typedef __attribute__((ext_vector_type(16))) _Float16 v16h;
typedef __attribute__((ext_vector_type(8)))  _Float16 v8h;
typedef __attribute__((ext_vector_type(8)))  float    v8f;
typedef __attribute__((ext_vector_type(4)))  unsigned int v4u;
typedef __attribute__((ext_vector_type(4)))  int      v4i;
typedef __attribute__((ext_vector_type(8)))  int      v8i;

#define DIM   128
#define NTOK  64
#define NHEAD 4
#define HD    32
#define NWIN  256
#define NBLK  8192
#define SCALE 0.17677669529663687f

// workspace layout (bytes)
#define WS_WQ_OFF   0        // 3*DIM*DIM halves = 98304 B
#define WS_WP_OFF   98304    // DIM*DIM halves   = 32768 B
#define WS_BIAS_OFF 131072   // NHEAD*64*64 f32  = 65536 B

// dynamic LDS layout (bytes), total 131072:
//  [0,65536)        : S  f32 [4][64][64]   (overlays xs f16[64][128] and ps f16[4][64][64])
//  [65536,81920)    : qs f16 [4][64][32]   (overlays oat f16[64][128] in phase 5)
//  [81920,98304)    : ks f16 [4][64][32]
//  [98304,114688)   : vt f16 [4][32][64]   (V transposed)
//  [114688,131072)  : maskL f32 [64][64]   (TDM-loaded)
#define LDS_MASK_OFF 114688
#define SMEM_BYTES   131072

// ---------- fragment loaders (CDNA5 f16 WMMA layouts, wave32) ----------
// A 16x32: lane<16 holds row(lane) K{0..7,16..23}; lane>=16 row(lane-16) K{8..15,24..31}
static __device__ inline v16h load_A(const h16* base, int ld, int lane) {
  const h16* p = base + (lane & 15) * ld + ((lane & 16) ? 8 : 0);
  v8h lo = *(const v8h*)p;
  v8h hi = *(const v8h*)(p + 16);
  v16h a;
#pragma unroll
  for (int i = 0; i < 8; ++i) { a[i] = lo[i]; a[8 + i] = hi[i]; }
  return a;
}
// B 32x16 from "transposed source": src row n holds B[:,n] contiguous over K.
// lane<16 holds col(lane) K{0..7,16..23}; lane>=16 col(lane-16) K{8..15,24..31}
static __device__ inline v16h load_Bt(const h16* base, int ld, int lane) {
  const h16* p = base + (lane & 15) * ld + ((lane & 16) ? 8 : 0);
  v8h lo = *(const v8h*)p;
  v8h hi = *(const v8h*)(p + 16);
  v16h b;
#pragma unroll
  for (int i = 0; i < 8; ++i) { b[i] = lo[i]; b[8 + i] = hi[i]; }
  return b;
}
static __device__ inline v8f wmma_f16(v16h a, v16h b, v8f c) {
  return __builtin_amdgcn_wmma_f32_16x16x32_f16(false, a, false, b, (short)0, c,
                                                false, false);
}

// ---------- TDM: 1-D tile of `elems` f32 from global -> LDS ----------
static __device__ inline void tdm_load_f32(const float* gptr, unsigned lds_off,
                                           unsigned elems) {
  unsigned long long ga = (unsigned long long)gptr;
  v4u g0;
  g0[0] = 1u;                                            // count=1
  g0[1] = lds_off;                                       // lds_addr (bytes)
  g0[2] = (unsigned)ga;                                  // global_addr[31:0]
  g0[3] = (unsigned)((ga >> 32) & 0x1FFFFFFu) | 0x80000000u; // addr[56:32] | type=2
  v8i g1;
  g1[0] = (int)(2u << 16);                               // data_size = 4B
  g1[1] = (int)((elems & 0xFFFFu) << 16);                // tensor_dim0 low16
  g1[2] = (int)((elems >> 16) | (1u << 16));             // tensor_dim0 hi16 | tensor_dim1=1
  g1[3] = (int)(elems << 16);                            // tile_dim0
  g1[4] = 0;                                             // tile_dim1=0, tile_dim2=0
  g1[5] = (int)elems;                                    // tensor_dim0_stride low32
  g1[6] = 0;
  g1[7] = 0;
#if __clang_major__ >= 23
  __builtin_amdgcn_tensor_load_to_lds(g0, g1, (v4i)0, (v4i)0, (v8i)0, 0);
#else
  __builtin_amdgcn_tensor_load_to_lds(g0, g1, (v4i)0, (v4i)0, 0);
#endif
}

// ---------- prep: weights fp32->f16, bias gather ----------
__global__ __launch_bounds__(256)
void prep_kernel(const float* __restrict__ qkv_w, const float* __restrict__ proj_w,
                 const float* __restrict__ bias_table, const int* __restrict__ rel_index,
                 h16* __restrict__ wq, h16* __restrict__ wp, float* __restrict__ biasf) {
  int idx = blockIdx.x * 256 + threadIdx.x;
  if (idx < 3 * DIM * DIM) {
    wq[idx] = (h16)qkv_w[idx];
  } else if (idx < 3 * DIM * DIM + DIM * DIM) {
    int j = idx - 3 * DIM * DIM;
    wp[j] = (h16)proj_w[j];
  } else if (idx < 3 * DIM * DIM + DIM * DIM + NHEAD * NTOK * NTOK) {
    int j = idx - (3 * DIM * DIM + DIM * DIM);
    int h = j >> 12;          // / 4096
    int rc = j & 4095;        // i*64+j within [64][64]
    biasf[j] = bias_table[rel_index[rc] * NHEAD + h];
  }
}

// ---------- main: one workgroup per window ----------
__global__ __launch_bounds__(256)
void win_attn_kernel(const float* __restrict__ x, const float* __restrict__ mask,
                     const h16* __restrict__ wq, const float* __restrict__ bq,
                     const h16* __restrict__ wp, const float* __restrict__ bp,
                     const float* __restrict__ biasf, float* __restrict__ out) {
  extern __shared__ char smem[];
  float* S    = (float*)smem;                    // [4][64][64]
  h16*   xs   = (h16*)smem;                      // [64][128]
  h16*   ps   = (h16*)smem;                      // [4][64][64]
  h16*   qs   = (h16*)(smem + 65536);            // [4][64][32]
  h16*   oat  = (h16*)(smem + 65536);            // [64][128] (after qs dead)
  h16*   ks   = (h16*)(smem + 81920);            // [4][64][32]
  h16*   vt   = (h16*)(smem + 98304);            // [4][32][64]
  float* maskL= (float*)(smem + LDS_MASK_OFF);   // [64][64]

  const int tid  = threadIdx.x;
  const int lane = tid & 31;
  const int wave = tid >> 5;
  const int blk  = blockIdx.x;
  const int widx = blk & (NWIN - 1);
  const int colL = lane & 15;
  const int rowL = (lane & 16) ? 8 : 0;

  // ---- async: TDM mask[widx] (16KB) -> LDS, overlapped with phases 0-1 ----
  if (wave == 0) {
    tdm_load_f32(mask + (size_t)widx * (NTOK * NTOK), LDS_MASK_OFF, NTOK * NTOK);
  }

  // ---- phase 0: x -> f16 LDS ----
  const float* xg = x + (size_t)blk * (NTOK * DIM);
#pragma unroll
  for (int i = tid; i < NTOK * DIM / 4; i += 256) {
    float4 f = ((const float4*)xg)[i];
    xs[4 * i + 0] = (h16)f.x; xs[4 * i + 1] = (h16)f.y;
    xs[4 * i + 2] = (h16)f.z; xs[4 * i + 3] = (h16)f.w;
  }
  __syncthreads();

  // ---- phase 1: qkv = x @ Wqkv^T + b  (M=64,N=384,K=128) -> q,k,v(T) LDS ----
#pragma unroll
  for (int it = 0; it < 12; ++it) {
    int t  = wave * 12 + it;       // 96 tiles
    int mt = t & 3, nt = t >> 2;   // nt in [0,24)
    v8f c = {};
#pragma unroll
    for (int kk = 0; kk < 4; ++kk) {
      v16h a = load_A(xs + mt * 16 * DIM + kk * 32, DIM, lane);
      v16h b = load_Bt(wq + (nt * 16) * DIM + kk * 32, DIM, lane);
      c = wmma_f16(a, b, c);
    }
    int o0  = nt * 16;
    // sec / head are wave-uniform: force scalar branches (keep EXEC all-1s)
    int sec = __builtin_amdgcn_readfirstlane(o0 >> 7);   // 0=q 1=k 2=v
    int h   = __builtin_amdgcn_readfirstlane((o0 & 127) >> 5);
    int dcol = ((o0 & 127) & 31) + colL;
    float bb = bq[o0 + colL];
    int rb   = mt * 16 + rowL;
    if (sec == 0) {
#pragma unroll
      for (int i = 0; i < 8; ++i) qs[h * 2048 + (rb + i) * HD + dcol] = (h16)(c[i] + bb);
    } else if (sec == 1) {
#pragma unroll
      for (int i = 0; i < 8; ++i) ks[h * 2048 + (rb + i) * HD + dcol] = (h16)(c[i] + bb);
    } else {
#pragma unroll
      for (int i = 0; i < 8; ++i) vt[h * 2048 + dcol * NTOK + (rb + i)] = (h16)(c[i] + bb);
    }
  }
  if (wave == 0) __builtin_amdgcn_s_wait_tensorcnt(0);   // mask tile landed
  __syncthreads();

  // ---- phase 2: S = scale*Q@K^T + bias + mask (per head, K=32) ----
#pragma unroll
  for (int it = 0; it < 8; ++it) {
    int t = wave * 8 + it;         // 64 tiles
    int h = t >> 4, rem = t & 15;
    int mt = rem >> 2, nt = rem & 3;
    v16h a = load_A(qs + h * 2048 + mt * 16 * HD, HD, lane);
    v16h b = load_Bt(ks + h * 2048 + (nt * 16) * HD, HD, lane);
    v8f c = {};
    c = wmma_f16(a, b, c);
    int col = nt * 16 + colL;
    int rb  = mt * 16 + rowL;
    const float* bptr = biasf + h * 4096 + rb * NTOK + col;   // global (L2-hot)
    const float* mptr = maskL + rb * NTOK + col;              // LDS (TDM-loaded)
#pragma unroll
    for (int i = 0; i < 8; ++i)
      S[h * 4096 + (rb + i) * NTOK + col] = c[i] * SCALE + bptr[i * NTOK] + mptr[i * NTOK];
  }
  __syncthreads();

  // ---- phase 3: softmax over last dim; one thread per (head,row) ----
  {
    int h = tid >> 6, r = tid & 63;
    const float* row = S + h * 4096 + r * NTOK;
    float vbuf[NTOK];
#pragma unroll
    for (int i = 0; i < NTOK / 4; ++i) {
      float4 f = ((const float4*)row)[i];
      vbuf[4 * i + 0] = f.x; vbuf[4 * i + 1] = f.y;
      vbuf[4 * i + 2] = f.z; vbuf[4 * i + 3] = f.w;
    }
    float m = vbuf[0];
#pragma unroll
    for (int i = 1; i < NTOK; ++i) m = fmaxf(m, vbuf[i]);
    float s = 0.f;
#pragma unroll
    for (int i = 0; i < NTOK; ++i) { vbuf[i] = __expf(vbuf[i] - m); s += vbuf[i]; }
    float inv = 1.f / s;
    __syncthreads();                       // all rows buffered before f16 overwrite
#pragma unroll
    for (int i = 0; i < NTOK; ++i) ps[h * 4096 + r * NTOK + i] = (h16)(vbuf[i] * inv);
  }
  __syncthreads();

  // ---- phase 4: O = P @ V (per head: M=64,K=64,N=32) -> oat f16 [64][128] ----
#pragma unroll
  for (int it = 0; it < 4; ++it) {
    int t = wave * 4 + it;         // 32 tiles
    int h = t >> 3, rem = t & 7;
    int mt = rem >> 1, nt = rem & 1;
    v8f c = {};
#pragma unroll
    for (int kk = 0; kk < 2; ++kk) {
      v16h a = load_A(ps + h * 4096 + mt * 16 * NTOK + kk * 32, NTOK, lane);
      v16h b = load_Bt(vt + h * 2048 + (nt * 16) * NTOK + kk * 32, NTOK, lane);
      c = wmma_f16(a, b, c);
    }
    int col = h * HD + nt * 16 + colL;
    int rb  = mt * 16 + rowL;
#pragma unroll
    for (int i = 0; i < 8; ++i) oat[(rb + i) * DIM + col] = (h16)c[i];
  }
  __syncthreads();

  // ---- phase 5: out = O @ Wp^T + bp (M=64,N=128,K=128) -> global f32 ----
  float* og = out + (size_t)blk * (NTOK * DIM);
#pragma unroll
  for (int it = 0; it < 4; ++it) {
    int t = wave * 4 + it;         // 32 tiles
    int mt = t & 3, nt = t >> 2;   // nt in [0,8)
    v8f c = {};
#pragma unroll
    for (int kk = 0; kk < 4; ++kk) {
      v16h a = load_A(oat + mt * 16 * DIM + kk * 32, DIM, lane);
      v16h b = load_Bt(wp + (nt * 16) * DIM + kk * 32, DIM, lane);
      c = wmma_f16(a, b, c);
    }
    int col = nt * 16 + colL;
    float bb = bp[col];
    int rb = mt * 16 + rowL;
#pragma unroll
    for (int i = 0; i < 8; ++i) og[(rb + i) * DIM + col] = c[i] + bb;
  }
}

extern "C" void kernel_launch(void* const* d_in, const int* in_sizes, int n_in,
                              void* d_out, int out_size, void* d_ws, size_t ws_size,
                              hipStream_t stream) {
  (void)in_sizes; (void)n_in; (void)out_size; (void)ws_size;
  const float* x          = (const float*)d_in[0];
  const float* mask       = (const float*)d_in[1];
  const float* qkv_w      = (const float*)d_in[2];
  const float* qkv_b      = (const float*)d_in[3];
  const float* proj_w     = (const float*)d_in[4];
  const float* proj_b     = (const float*)d_in[5];
  const float* bias_table = (const float*)d_in[6];
  const int*   rel_index  = (const int*)d_in[7];

  char* ws = (char*)d_ws;
  h16*   wq    = (h16*)(ws + WS_WQ_OFF);
  h16*   wp    = (h16*)(ws + WS_WP_OFF);
  float* biasf = (float*)(ws + WS_BIAS_OFF);

  // 3*128*128 + 128*128 + 4*64*64 = 81920 elements -> 320 blocks of 256
  prep_kernel<<<320, 256, 0, stream>>>(qkv_w, proj_w, bias_table, rel_index,
                                       wq, wp, biasf);
  win_attn_kernel<<<NBLK, 256, SMEM_BYTES, stream>>>(x, mask, wq, qkv_b,
                                                     wp, proj_b, biasf,
                                                     (float*)d_out);
}
